// KGSSMEncoder_24404004176099
// MI455X (gfx1250) — matrix operand order repaired and checked by reading, not compiled, not run on hardware
//
#include <hip/hip_runtime.h>

#define DIM     256
#define TOK     3
#define NLAYERS 4
#define MROWS   32      // batch rows per block
#define WAVES   4
#define NTPW    4       // 16-col N tiles per wave  (4 waves * 4 = 16 tiles = 256 cols)
#define MTPW    2       // 16-row M tiles per wave  (2 * 16 = 32 rows)

typedef __attribute__((ext_vector_type(16))) __bf16 v16bf;
typedef __attribute__((ext_vector_type(8)))  float  v8f;

union ABU { uint4 u[2]; v16bf v; };

__device__ __forceinline__ unsigned short f2bf(float f) {
    unsigned int u = __float_as_uint(f);
    u += 0x7FFFu + ((u >> 16) & 1u);          // round-to-nearest-even
    return (unsigned short)(u >> 16);
}
__device__ __forceinline__ float bf2f(unsigned short h) {
    return __uint_as_float(((unsigned int)h) << 16);
}

__device__ __forceinline__ void zero_frags(v8f a[MTPW][NTPW]) {
#pragma unroll
    for (int mt = 0; mt < MTPW; ++mt)
#pragma unroll
        for (int j = 0; j < NTPW; ++j)
#pragma unroll
            for (int i = 0; i < 8; ++i) a[mt][j][i] = 0.0f;
}

// acc[mt][j] += X(tile mt of aLds) @ W(ntile wave*4+j)
// aLds: [MROWS][DIM] bf16 row-major activations in LDS
// wG:   [DIM][DIM]   bf16 "wT" layout in global (row n holds W[:,n], i.e. out = X @ W)
__device__ __forceinline__ void mm_acc(const unsigned short* aLds,
                                       const unsigned short* __restrict__ wG,
                                       v8f acc[MTPW][NTPW], int lane, int wave)
{
    const int sel = lane >> 4;       // half-wave select
    const int lr  = lane & 15;
#pragma unroll
    for (int kt = 0; kt < 8; ++kt) {             // K = 256 = 8 * 32
        ABU a[MTPW];
#pragma unroll
        for (int mt = 0; mt < MTPW; ++mt) {
            const unsigned short* ap = aLds + (mt*16 + lr)*DIM + kt*32 + sel*8;
            a[mt].u[0] = *(const uint4*)ap;          // K 0..7   (or 8..15)
            a[mt].u[1] = *(const uint4*)(ap + 16);   // K 16..23 (or 24..31)
        }
#pragma unroll
        for (int j = 0; j < NTPW; ++j) {
            const int n = (wave*NTPW + j)*16 + lr;
            ABU b;
            const unsigned short* bp = wG + n*DIM + kt*32 + sel*8;
            b.u[0] = *(const uint4*)bp;              // K 0..7   (or 8..15)
            b.u[1] = *(const uint4*)(bp + 16);       // K 16..23 (or 24..31)
#pragma unroll
            for (int mt = 0; mt < MTPW; ++mt)
                acc[mt][j] = __builtin_amdgcn_wmma_f32_16x16x32_bf16(
                    false, a[mt].v, false, b.v, (short)0, acc[mt][j], false, false);
        }
    }
}

// Write fp32 fragments to LDS as bf16 (row-major [MROWS][DIM]).
__device__ __forceinline__ void store_frags_bf16(const v8f acc[MTPW][NTPW],
                                                 unsigned short* dst, int lane, int wave)
{
    const int sel = lane >> 4, lr = lane & 15;
#pragma unroll
    for (int mt = 0; mt < MTPW; ++mt)
#pragma unroll
        for (int j = 0; j < NTPW; ++j) {
            const int n = (wave*NTPW + j)*16 + lr;
#pragma unroll
            for (int v = 0; v < 8; ++v) {
                const int m = mt*16 + v + sel*8;     // C/D layout: lanes>=16 hold M=v+8
                dst[m*DIM + n] = f2bf(acc[mt][j][v]);
            }
        }
}

// LayerNorm over the row dimension of y (frags in registers), residual-add into act (bf16 LDS).
__device__ __forceinline__ void ln_residual(const v8f y[MTPW][NTPW],
                                            unsigned short* act,          // [MROWS][DIM]
                                            const float* __restrict__ gamma,
                                            const float* __restrict__ beta,
                                            float* red, float* mu_s, float* rs_s,
                                            int lane, int wave, int tid)
{
    const int sel = lane >> 4, lr = lane & 15;
    // per-wave partial sums over this wave's 64 columns, reduced across the 16-lane group
#pragma unroll
    for (int mt = 0; mt < MTPW; ++mt)
#pragma unroll
        for (int v = 0; v < 8; ++v) {
            float s = 0.f, s2 = 0.f;
#pragma unroll
            for (int j = 0; j < NTPW; ++j) { float x = y[mt][j][v]; s += x; s2 += x*x; }
#pragma unroll
            for (int m = 1; m < 16; m <<= 1) {
                s  += __shfl_xor(s,  m, 32);
                s2 += __shfl_xor(s2, m, 32);
            }
            if (lr == 0) {
                const int row = mt*16 + v + sel*8;
                red[(row*WAVES + wave)*2 + 0] = s;
                red[(row*WAVES + wave)*2 + 1] = s2;
            }
        }
    __syncthreads();
    if (tid < MROWS) {
        float s = 0.f, s2 = 0.f;
#pragma unroll
        for (int w = 0; w < WAVES; ++w) {
            s  += red[(tid*WAVES + w)*2 + 0];
            s2 += red[(tid*WAVES + w)*2 + 1];
        }
        const float mu  = s  * (1.0f / DIM);
        const float var = s2 * (1.0f / DIM) - mu*mu;
        mu_s[tid] = mu;
        rs_s[tid] = __frsqrt_rn(var + 1e-5f);
    }
    __syncthreads();
#pragma unroll
    for (int j = 0; j < NTPW; ++j) {
        const int n = (wave*NTPW + j)*16 + lr;
        const float g = gamma[n], b = beta[n];
#pragma unroll
        for (int mt = 0; mt < MTPW; ++mt)
#pragma unroll
            for (int v = 0; v < 8; ++v) {
                const int m  = mt*16 + v + sel*8;
                const float yn = (y[mt][j][v] - mu_s[m]) * rs_s[m] * g + b;
                const float nv = bf2f(act[m*DIM + n]) + yn;     // residual
                act[m*DIM + n] = f2bf(nv);
            }
    }
    __syncthreads();
}

// ---- weight prep: fp32 -> bf16, transposed to wT[n][k] layout -------------
// wbf layout: mats 0..11 = layer l {A,B,C} transposed; mat 12 = fc_w (already wT).
__global__ __launch_bounds__(256) void kg_prep(const float* __restrict__ A,
                                               const float* __restrict__ B,
                                               const float* __restrict__ C,
                                               const float* __restrict__ fcw,
                                               unsigned short* __restrict__ wbf)
{
    const int i = blockIdx.x * 256 + threadIdx.x;   // 13 * 65536 total
    const int mat = i >> 16;
    const int r   = i & 65535;
    float v;
    if (mat < 12) {
        const int l = mat / 3, which = mat % 3;
        const int n = r >> 8, k = r & 255;
        const float* src = (which == 0) ? A : (which == 1) ? B : C;
        v = src[l*DIM*DIM + k*DIM + n];             // transpose: wT[n][k] = M[k][n]
    } else {
        v = fcw[r];                                  // out = last @ fc_w.T  ->  wT = fc_w
    }
    wbf[i] = f2bf(v);
}

// ---- main fused kernel ----------------------------------------------------
// No min-occupancy bound: let the compiler keep weight fragments register-
// resident (no scratch spills). LN dead-time is hidden by issuing the next
// token's x@B WMMAs *before* the LN (XDL co-executes with VALU).
__global__ __launch_bounds__(128) void kg_ssm_main(
    const int* __restrict__ head, const int* __restrict__ rel, const int* __restrict__ tail,
    const float* __restrict__ ent, const float* __restrict__ relE,
    const float* __restrict__ gamma, const float* __restrict__ beta,
    const float* __restrict__ fcb,
    const unsigned short* __restrict__ wbf,
    float* __restrict__ out)
{
    __shared__ unsigned short actB[TOK][MROWS][DIM];   // 48 KB bf16 activations
    __shared__ unsigned short hB[MROWS][DIM];          // 16 KB bf16 SSM state
    __shared__ float red[MROWS*WAVES*2];
    __shared__ float mu_s[MROWS], rs_s[MROWS];

    const int tid  = threadIdx.x;
    const int lane = tid & 31, wave = tid >> 5;
    const int g0   = blockIdx.x * MROWS;

    // gather embeddings -> bf16 LDS (vectorized: float4 loads, 8B packed stores)
    for (int idx = tid; idx < TOK*MROWS*(DIM/4); idx += 128) {
        const int t = idx / (MROWS*(DIM/4));
        const int m = (idx / (DIM/4)) % MROWS;
        const int k = (idx & (DIM/4 - 1)) * 4;
        int id; const float* src;
        if      (t == 0) { id = head[g0 + m]; src = ent;  }
        else if (t == 1) { id = rel [g0 + m]; src = relE; }
        else             { id = tail[g0 + m]; src = ent;  }
        const float4 f = *(const float4*)(src + (size_t)id * DIM + k);
        ushort4 h4;
        h4.x = f2bf(f.x); h4.y = f2bf(f.y); h4.z = f2bf(f.z); h4.w = f2bf(f.w);
        *(ushort4*)&actB[t][m][k] = h4;
    }
    __syncthreads();

    v8f accP[MTPW][NTPW];   // pipeline accumulator: holds x_t@B, then becomes h_t
    v8f accY[MTPW][NTPW];   // y_t accumulator (live through LN)

    // prologue: accP = x_0 @ B_0
    zero_frags(accP);
    mm_acc(&actB[0][0][0], wbf + (size_t)(0*3 + 1)*DIM*DIM, accP, lane, wave);

#pragma unroll 1
    for (int l = 0; l < NLAYERS; ++l) {
        const unsigned short* Aw = wbf + (size_t)(l*3 + 0)*DIM*DIM;
        const unsigned short* Cw = wbf + (size_t)(l*3 + 2)*DIM*DIM;
        const float* gl = gamma + l*DIM;
        const float* bl = beta  + l*DIM;

#pragma unroll
        for (int t = 0; t < TOK; ++t) {
            // accP currently = x_t @ B;  h_t = accP (+ h_{t-1} @ A for t > 0)
            if (t > 0) mm_acc(&hB[0][0], Aw, accP, lane, wave);
            __syncthreads();                       // all reads of old hB done
            store_frags_bf16(accP, &hB[0][0], lane, wave);
            __syncthreads();
            // y_t = h_t @ C
            zero_frags(accY);
            mm_acc(&hB[0][0], Cw, accY, lane, wave);
            // prefetch next token's x@B: issue its WMMAs now, they drain in the
            // matrix pipe while the LN below runs on the VALU.
            if (!(l == NLAYERS-1 && t == TOK-1)) {
                const unsigned short* nextB = (t < TOK-1)
                    ? wbf + (size_t)(l*3 + 1)*DIM*DIM          // same layer B
                    : wbf + (size_t)((l+1)*3 + 1)*DIM*DIM;     // next layer B
                const unsigned short* nextX = (t < TOK-1)
                    ? &actB[t+1][0][0]
                    : &actB[0][0][0];      // post-LN value from this layer's t=0
                zero_frags(accP);
                mm_acc(nextX, nextB, accP, lane, wave);
            }
            // seq[t] += LayerNorm(y_t)   (VALU work, overlaps the WMMAs above)
            ln_residual(accY, &actB[t][0][0], gl, bl, red, mu_s, rs_s, lane, wave, tid);
        }
    }

    // epilogue: out = seq[:, 2, :] @ fc_w.T + fc_b  (needs post-LN actB[2])
    zero_frags(accP);
    mm_acc(&actB[2][0][0], wbf + (size_t)12*DIM*DIM, accP, lane, wave);
    const int sel = lane >> 4, lr = lane & 15;
#pragma unroll
    for (int j = 0; j < NTPW; ++j) {
        const int n = (wave*NTPW + j)*16 + lr;
        const float b = fcb[n];
#pragma unroll
        for (int mt = 0; mt < MTPW; ++mt)
#pragma unroll
            for (int v = 0; v < 8; ++v) {
                const int m = mt*16 + v + sel*8;
                out[(size_t)(g0 + m)*DIM + n] = accP[mt][j][v] + b;
            }
    }
}

extern "C" void kernel_launch(void* const* d_in, const int* in_sizes, int n_in,
                              void* d_out, int out_size, void* d_ws, size_t ws_size,
                              hipStream_t stream)
{
    const int*   head  = (const int*)  d_in[0];
    const int*   rel   = (const int*)  d_in[1];
    const int*   tail  = (const int*)  d_in[2];
    const float* ent   = (const float*)d_in[3];
    const float* relE  = (const float*)d_in[4];
    const float* A     = (const float*)d_in[5];
    const float* B     = (const float*)d_in[6];
    const float* C     = (const float*)d_in[7];
    const float* gamma = (const float*)d_in[8];
    const float* beta  = (const float*)d_in[9];
    const float* fcw   = (const float*)d_in[10];
    const float* fcb   = (const float*)d_in[11];

    unsigned short* wbf = (unsigned short*)d_ws;    // 13 * 65536 bf16 = 1.7 MB

    kg_prep<<<(13*65536)/256, 256, 0, stream>>>(A, B, C, fcw, wbf);

    const int batch = in_sizes[0];
    kg_ssm_main<<<batch / MROWS, 128, 0, stream>>>(
        head, rel, tail, ent, relE, gamma, beta, fcb, wbf, (float*)d_out);
}
